// TransConvLayer_8598524527213
// MI455X (gfx1250) — matrix operand order repaired and checked by reading, not compiled, not run on hardware
//
#include <hip/hip_runtime.h>

#define NN 20000   // nodes
#define DD 256     // feature dim (DIN == DOUT)
#define HH 8       // heads

typedef __attribute__((ext_vector_type(16))) __bf16 v16bf;
typedef __attribute__((ext_vector_type(8)))  float  v8f;

// ---------- scalar bf16 helpers ----------
__device__ __forceinline__ unsigned short f2bf(float f) {
  union { float f; unsigned u; } v; v.f = f;
  unsigned r = v.u + 0x7FFFu + ((v.u >> 16) & 1u);   // round-to-nearest-even
  return (unsigned short)(r >> 16);
}
__device__ __forceinline__ float bf2f(unsigned short u) {
  union { unsigned u; float f; } v; v.u = ((unsigned)u) << 16;
  return v.f;
}
__device__ __forceinline__ __bf16 us2bf(unsigned short u) {
  union { unsigned short u; __bf16 h; } v; v.u = u; return v.h;
}

// ---------- WMMA fragment builders (wave32, contiguous k = 16*half + s in both A and B) ----------
// A fragment, 16x32 bf16, from row-major tile A[m][k] with leading dim lda.
// Lane m reads 32 contiguous bytes (k = 16*half + s) -> ds_load_b128 / global b128.
__device__ __forceinline__ v16bf load_afrag(const unsigned short* __restrict__ A,
                                            int lda, int kbase) {
  const int lane = threadIdx.x & 31;
  const int half = lane >> 4;
  const int m    = lane & 15;
  v16bf a;
#pragma unroll
  for (int j = 0; j < 8; ++j) {
    int kb = kbase + half * 16 + 2 * j;
    a[2 * j]     = us2bf(A[m * lda + kb]);
    a[2 * j + 1] = us2bf(A[m * lda + kb + 1]);
  }
  return a;
}
// B fragment, 32x16 bf16, source layout P[n][k] (pairs along k contiguous).
// P must already point at column n0's row. Lane reads 32 contiguous bytes.
__device__ __forceinline__ v16bf load_bfrag_kmajor(const unsigned short* __restrict__ P,
                                                   int ldn, int kbase) {
  const int lane = threadIdx.x & 31;
  const int half = lane >> 4;
  const int n    = lane & 15;
  v16bf b;
#pragma unroll
  for (int j = 0; j < 8; ++j) {
    int kb = kbase + half * 16 + 2 * j;
    b[2 * j]     = us2bf(P[n * ldn + kb]);
    b[2 * j + 1] = us2bf(P[n * ldn + kb + 1]);
  }
  return b;
}

__device__ __forceinline__ v8f wmma_bf16(v16bf a, v16bf b, v8f c) {
  return __builtin_amdgcn_wmma_f32_16x16x32_bf16(false, a, false, b, (short)0, c,
                                                 false, false);
}
__device__ __forceinline__ v8f vzero8() {
  v8f z = {0.f, 0.f, 0.f, 0.f, 0.f, 0.f, 0.f, 0.f};
  return z;
}

// ---------- utility kernels ----------
__global__ void cvt_f32_bf16_kernel(const float* __restrict__ s,
                                    unsigned short* __restrict__ d, int n) {
  int i = blockIdx.x * 256 + threadIdx.x;
  if (i < n) d[i] = f2bf(s[i]);
}
__global__ void zero_f32_kernel(float* __restrict__ p, int n) {
  int i = blockIdx.x * 256 + threadIdx.x;
  if (i < n) p[i] = 0.f;
}

// ---------- stage 1: per-head QKV projection + phi ----------
// grid (1250, 8, 3) block 128. sel: 0=q(->phi_q) 1=k(->phi_k) 2=v(->vbuf)
__global__ void qkv_phi_kernel(const unsigned short* __restrict__ qbf,
                               const unsigned short* __restrict__ sbf,
                               const unsigned short* __restrict__ Wq,
                               const unsigned short* __restrict__ Wk,
                               const unsigned short* __restrict__ Wv,
                               const float* __restrict__ bq,
                               const float* __restrict__ bk,
                               const float* __restrict__ bv,
                               const float* __restrict__ norm_scale,
                               unsigned short* __restrict__ phiq,
                               unsigned short* __restrict__ phik,
                               unsigned short* __restrict__ vbuf) {
  const int rb  = blockIdx.x;
  const int h   = blockIdx.y;
  const int sel = blockIdx.z;
  const int tid = threadIdx.x;              // 0..127 (4 waves)
  const int wave = tid >> 5, lane = tid & 31, half = lane >> 4, nl = lane & 15;

  const unsigned short* X = (sel == 0) ? qbf : sbf;
  const unsigned short* W = (sel == 0) ? Wq : ((sel == 1) ? Wk : Wv);
  const float*          B = (sel == 0) ? bq : ((sel == 1) ? bk : bv);

  __shared__ __align__(16) unsigned short Atile[16 * 256];  // 8 KB
  __shared__ __align__(16) float Xt[16 * 256];              // 16 KB
  __shared__ float s2[16], s4[16];

  // stage A tile (16 rows x 256 k) from global, contiguous dwords
  {
    const unsigned* src = (const unsigned*)(X + (size_t)rb * 16 * 256);
    unsigned* dst = (unsigned*)Atile;
    for (int i = tid; i < 2048; i += 128) dst[i] = src[i];
  }
  if (tid < 16) { s2[tid] = 0.f; s4[tid] = 0.f; }
  __syncthreads();

  v8f acc[4] = {vzero8(), vzero8(), vzero8(), vzero8()};
  for (int kt = 0; kt < 8; ++kt) {
    v16bf a = load_afrag(Atile, 256, kt * 32);
#pragma unroll
    for (int t = 0; t < 4; ++t) {
      int n0 = (wave * 4 + t) * 16;
      v16bf b = load_bfrag_kmajor(W + ((size_t)h * 256 + n0) * 256, 256, kt * 32);
      acc[t] = wmma_bf16(a, b, acc[t]);
    }
  }

  const float inv_ns = 1.0f / fabsf(norm_scale[0]);

  if (sel == 2) {
#pragma unroll
    for (int t = 0; t < 4; ++t) {
      int n0 = (wave * 4 + t) * 16;
#pragma unroll
      for (int r = 0; r < 8; ++r) {
        int row = r + half * 8, col = n0 + nl;
        float val = acc[t][r] + B[h * 256 + col];
        vbuf[(((size_t)(rb * 16 + row) * HH + h) << 8) + col] = f2bf(val);
      }
    }
  } else {
    // x = (relu(y)+eps)/|ns|, store to LDS for row-norm reduction
#pragma unroll
    for (int t = 0; t < 4; ++t) {
      int n0 = (wave * 4 + t) * 16;
#pragma unroll
      for (int r = 0; r < 8; ++r) {
        int row = r + half * 8, col = n0 + nl;
        float val = acc[t][r] + B[h * 256 + col];
        float x = (fmaxf(val, 0.f) + 1e-6f) * inv_ns;
        Xt[row * 256 + col] = x;
      }
    }
    __syncthreads();
    {  // 8 threads per row, 32 cols each: sum x^2 and x^4
      int row = tid >> 3, c0 = (tid & 7) * 32;
      float p2 = 0.f, p4 = 0.f;
      for (int i = 0; i < 32; ++i) {
        float x = Xt[row * 256 + c0 + i];
        float xx = x * x;
        p2 += xx; p4 += xx * xx;
      }
      atomicAdd(&s2[row], p2);
      atomicAdd(&s4[row], p4);
    }
    __syncthreads();
    unsigned short* outp = (sel == 0) ? phiq : phik;
#pragma unroll
    for (int t = 0; t < 4; ++t) {
      int n0 = (wave * 4 + t) * 16;
#pragma unroll
      for (int r = 0; r < 8; ++r) {
        int row = r + half * 8, col = n0 + nl;
        float x = Xt[row * 256 + col];
        float sc = sqrtf(s2[row]) * rsqrtf(s4[row]);  // ||x|| / ||x^2||
        outp[(((size_t)(rb * 16 + row) * HH + h) << 8) + col] = f2bf(sc * x * x);
      }
    }
  }
}

// ---------- stage 2a: vbar = mean_h v ----------
__global__ void vbar_kernel(const unsigned short* __restrict__ vbuf,
                            unsigned short* __restrict__ vbar) {
  int idx = blockIdx.x * 256 + threadIdx.x;  // grid 20000 -> n,d exact
  int n = idx >> 8, d = idx & 255;
  float s = 0.f;
#pragma unroll
  for (int h = 0; h < HH; ++h) s += bf2f(vbuf[(((size_t)n * HH + h) << 8) + d]);
  vbar[(size_t)n * 256 + d] = f2bf(s * 0.125f);
}

// ---------- stage 2b: k_sum[h][m] = sum_n phi_k ----------
__global__ void ksum_kernel(const unsigned short* __restrict__ phik,
                            float* __restrict__ ksum) {
  int h = blockIdx.x, chunk = blockIdx.y, m = threadIdx.x;
  float p = 0.f;
  int base = chunk * 500;
  for (int i = 0; i < 500; ++i) {
    size_t node = (size_t)(base + i);
    p += bf2f(phik[((node * HH + h) << 8) + m]);
  }
  atomicAdd(&ksum[h * 256 + m], p);
}

// ---------- stage 3: ktv[h] = phi_k^T @ v, split-K over nodes ----------
// grid (16 mtiles, 8 heads, 4 k-chunks) block 256 (8 waves, 2 n-tiles each).
// LDS tiles are staged K-minor (transposed) so every fragment read is a pair
// of aligned ds_load_b128; next step's global rows are prefetched.
__global__ void ktv_kernel(const unsigned short* __restrict__ phik,
                           const unsigned short* __restrict__ vbuf,
                           float* __restrict__ ktv) {
  const int m0 = blockIdx.x * 16;
  const int h  = blockIdx.y;
  const int z  = blockIdx.z;
  const int tid = threadIdx.x;
  const int wave = tid >> 5, lane = tid & 31, half = lane >> 4, nl = lane & 15;

  __shared__ __align__(16) unsigned short phitT[16 * 32];   // [m][k-node] 1 KB
  __shared__ __align__(16) unsigned short vtT[256 * 32];    // [d][k-node] 16 KB

  const int step0 = z * 157;
  const int rem   = 625 - step0;
  const int steps = (rem < 157) ? rem : 157;

  v8f acc0 = vzero8(), acc1 = vzero8();
  for (int s = 0; s < steps; ++s) {
    const size_t node0 = (size_t)(step0 + s) * 32;
    // stage phi_k^T tile: A[m][k=node] K-minor
    for (int i = tid; i < 512; i += 256) {
      int node = i >> 4, m = i & 15;
      phitT[m * 32 + node] = phik[(((node0 + node) * HH + h) << 8) + m0 + m];
    }
    // stage v tile transposed: vtT[d][node], dword global reads
    for (int i = tid; i < 4096; i += 256) {
      int node = i >> 7, dp = (i & 127);
      unsigned w = *(const unsigned*)&vbuf[(((node0 + node) * HH + h) << 8) + dp * 2];
      int d = dp * 2;
      vtT[d * 32 + node]       = (unsigned short)(w & 0xFFFFu);
      vtT[(d + 1) * 32 + node] = (unsigned short)(w >> 16);
    }
    // prefetch next step's rows while this step's stores are in flight
    if (s + 1 < steps) {
      const size_t nn0 = node0 + 32;
      if (tid < 128) {
        int node = tid >> 2, part = (tid & 3) * 64;  // 4x128B per 512B v row
        __builtin_prefetch(&vbuf[(((nn0 + node) * HH + h) << 8) + part], 0, 1);
      } else if (tid < 160) {
        int node = tid - 128;
        __builtin_prefetch(&phik[(((nn0 + node) * HH + h) << 8) + m0], 0, 1);
      }
    }
    __syncthreads();
    v16bf a  = load_afrag(phitT, 32, 0);
    v16bf b0 = load_bfrag_kmajor(vtT + (size_t)(wave * 32) * 32, 32, 0);
    v16bf b1 = load_bfrag_kmajor(vtT + (size_t)(wave * 32 + 16) * 32, 32, 0);
    acc0 = wmma_bf16(a, b0, acc0);
    acc1 = wmma_bf16(a, b1, acc1);
    __syncthreads();
  }
#pragma unroll
  for (int r = 0; r < 8; ++r) {
    int row = r + half * 8;
    atomicAdd(&ktv[h * 65536 + (m0 + row) * 256 + wave * 32 + nl], acc0[r]);
    atomicAdd(&ktv[h * 65536 + (m0 + row) * 256 + wave * 32 + 16 + nl], acc1[r]);
  }
}

// ---------- stage 3b: transpose-convert ktv f32[h][m][d] -> bf16[h][d][m] ----------
__global__ void ktvT_kernel(const float* __restrict__ ktv,
                            unsigned short* __restrict__ ktvT) {
  int flat = blockIdx.x * 256 + threadIdx.x;  // grid 2048 -> 524288 exact
  int h = flat >> 16, remv = flat & 65535, m = remv >> 8, d = remv & 255;
  ktvT[h * 65536 + d * 256 + m] = f2bf(ktv[flat]);
}

// ---------- stage 4: numerator/denominator + vss + head-mean + time ----------
// grid (1250) block 256 (8 waves, 2 n-tiles each)
__global__ void out_kernel(const unsigned short* __restrict__ phiq,
                           const unsigned short* __restrict__ vbar,
                           const unsigned short* __restrict__ ktvT,
                           const float* __restrict__ ksum,
                           const unsigned short* __restrict__ vmapbf,
                           const float* __restrict__ vmap_b,
                           float* __restrict__ out) {
  const int rb  = blockIdx.x;
  const int tid = threadIdx.x;
  const int wave = tid >> 5, lane = tid & 31, half = lane >> 4, nl = lane & 15;

  __shared__ __align__(16) unsigned short Atile[16 * 256];  // 8 KB
  __shared__ float accs[16 * 256];                          // 16 KB
  __shared__ float den[16];
  __shared__ float s2[16];

  for (int i = tid; i < 4096; i += 256) accs[i] = 0.f;

  // ---- vss = vbar @ vmap^T (+ vmap_b), already the head-mean contribution
  {
    const unsigned* src = (const unsigned*)(vbar + (size_t)rb * 16 * 256);
    unsigned* dst = (unsigned*)Atile;
    for (int i = tid; i < 2048; i += 256) dst[i] = src[i];
  }
  __syncthreads();
  {
    v8f a0 = vzero8(), a1 = vzero8();
    for (int kt = 0; kt < 8; ++kt) {
      v16bf a = load_afrag(Atile, 256, kt * 32);
      a0 = wmma_bf16(a, load_bfrag_kmajor(vmapbf + (size_t)(wave * 32) * 256, 256, kt * 32), a0);
      a1 = wmma_bf16(a, load_bfrag_kmajor(vmapbf + (size_t)(wave * 32 + 16) * 256, 256, kt * 32), a1);
    }
#pragma unroll
    for (int r = 0; r < 8; ++r) {
      int row = r + half * 8;
      int c0 = wave * 32 + nl, c1 = wave * 32 + 16 + nl;
      accs[row * 256 + c0] += a0[r] + vmap_b[c0];
      accs[row * 256 + c1] += a1[r] + vmap_b[c1];
    }
  }
  __syncthreads();

  // ---- per-head: numerator = phi_q @ ktv, scaled by 1/(H*(den+eps))
  for (int h = 0; h < HH; ++h) {
    for (int i = tid; i < 2048; i += 256) {
      int row = i >> 7, dd = (i & 127) * 2;
      *(unsigned*)&Atile[row * 256 + dd] =
          *(const unsigned*)&phiq[(((size_t)(rb * 16 + row) * HH + h) << 8) + dd];
    }
    if (tid < 16) den[tid] = 0.f;
    __syncthreads();
    {
      int row = tid >> 4, c0 = (tid & 15) * 16;
      float p = 0.f;
      for (int i = 0; i < 16; ++i)
        p += bf2f(Atile[row * 256 + c0 + i]) * ksum[h * 256 + c0 + i];
      atomicAdd(&den[row], p);
    }
    __syncthreads();

    v8f a0 = vzero8(), a1 = vzero8();
    const unsigned short* Bh = ktvT + (size_t)h * 65536;
    for (int kt = 0; kt < 8; ++kt) {
      v16bf a = load_afrag(Atile, 256, kt * 32);
      a0 = wmma_bf16(a, load_bfrag_kmajor(Bh + (size_t)(wave * 32) * 256, 256, kt * 32), a0);
      a1 = wmma_bf16(a, load_bfrag_kmajor(Bh + (size_t)(wave * 32 + 16) * 256, 256, kt * 32), a1);
    }
#pragma unroll
    for (int r = 0; r < 8; ++r) {
      int row = r + half * 8;
      float sc = 0.125f / (den[row] + 1e-6f);
      accs[row * 256 + wave * 32 + nl]      += a0[r] * sc;
      accs[row * 256 + wave * 32 + 16 + nl] += a1[r] * sc;
    }
    __syncthreads();
  }

  // ---- Lorentz time component + store
  if (tid < 16) s2[tid] = 0.f;
  __syncthreads();
  {
    int row = tid >> 4, c0 = (tid & 15) * 16;
    float p = 0.f;
    for (int i = 0; i < 16; ++i) {
      float x = accs[row * 256 + c0 + i];
      p += x * x;
    }
    atomicAdd(&s2[row], p);
  }
  __syncthreads();
  for (int i = tid; i < 4096; i += 256) {
    int row = i >> 8, col = i & 255;
    out[(size_t)(rb * 16 + row) * 257 + 1 + col] = accs[i];
  }
  if (tid < 16) out[(size_t)(rb * 16 + tid) * 257] = sqrtf(s2[tid] + 1.0f);
}

// ---------- host launch ----------
extern "C" void kernel_launch(void* const* d_in, const int* in_sizes, int n_in,
                              void* d_out, int out_size, void* d_ws, size_t ws_size,
                              hipStream_t stream) {
  (void)in_sizes; (void)n_in; (void)out_size; (void)ws_size;
  const float* q_in   = (const float*)d_in[0];
  const float* s_in   = (const float*)d_in[1];
  const float* Wq_w   = (const float*)d_in[2];
  const float* Wq_b   = (const float*)d_in[3];
  const float* Wk_w   = (const float*)d_in[4];
  const float* Wk_b   = (const float*)d_in[5];
  const float* Wv_w   = (const float*)d_in[6];
  const float* Wv_b   = (const float*)d_in[7];
  const float* nscale = (const float*)d_in[8];
  const float* vmap_w = (const float*)d_in[9];
  const float* vmap_b = (const float*)d_in[10];
  float* out = (float*)d_out;

  char* ws = (char*)d_ws;
  size_t off = 0;
  auto take = [&](size_t bytes) -> char* {
    char* p = ws + off;
    off += (bytes + 255) & ~(size_t)255;
    return p;
  };
  unsigned short* qbf    = (unsigned short*)take((size_t)NN * DD * 2);
  unsigned short* sbf    = (unsigned short*)take((size_t)NN * DD * 2);
  unsigned short* wqbf   = (unsigned short*)take((size_t)HH * DD * DD * 2);
  unsigned short* wkbf   = (unsigned short*)take((size_t)HH * DD * DD * 2);
  unsigned short* wvbf   = (unsigned short*)take((size_t)HH * DD * DD * 2);
  unsigned short* vmapbf = (unsigned short*)take((size_t)DD * DD * 2);
  unsigned short* phiq   = (unsigned short*)take((size_t)NN * HH * DD * 2);
  unsigned short* phik   = (unsigned short*)take((size_t)NN * HH * DD * 2);
  unsigned short* vbuf   = (unsigned short*)take((size_t)NN * HH * DD * 2);
  unsigned short* vbar   = (unsigned short*)take((size_t)NN * DD * 2);
  float*          ktvf   = (float*)take((size_t)HH * DD * DD * 4);
  unsigned short* ktvT   = (unsigned short*)take((size_t)HH * DD * DD * 2);
  float*          ksum   = (float*)take((size_t)HH * DD * 4);

  // 1) bf16 conversions
  cvt_f32_bf16_kernel<<<(NN * DD + 255) / 256, 256, 0, stream>>>(q_in, qbf, NN * DD);
  cvt_f32_bf16_kernel<<<(NN * DD + 255) / 256, 256, 0, stream>>>(s_in, sbf, NN * DD);
  cvt_f32_bf16_kernel<<<(HH * DD * DD + 255) / 256, 256, 0, stream>>>(Wq_w, wqbf, HH * DD * DD);
  cvt_f32_bf16_kernel<<<(HH * DD * DD + 255) / 256, 256, 0, stream>>>(Wk_w, wkbf, HH * DD * DD);
  cvt_f32_bf16_kernel<<<(HH * DD * DD + 255) / 256, 256, 0, stream>>>(Wv_w, wvbf, HH * DD * DD);
  cvt_f32_bf16_kernel<<<(DD * DD + 255) / 256, 256, 0, stream>>>(vmap_w, vmapbf, DD * DD);
  // zero f32 accumulators (ws is poisoned)
  zero_f32_kernel<<<(HH * DD * DD + 255) / 256, 256, 0, stream>>>(ktvf, HH * DD * DD);
  zero_f32_kernel<<<(HH * DD + 255) / 256, 256, 0, stream>>>(ksum, HH * DD);

  // 2) QKV projections + phi
  qkv_phi_kernel<<<dim3(NN / 16, HH, 3), 128, 0, stream>>>(
      qbf, sbf, wqbf, wkbf, wvbf, Wq_b, Wk_b, Wv_b, nscale, phiq, phik, vbuf);

  // 3) head-mean of v, k_sum
  vbar_kernel<<<NN, 256, 0, stream>>>(vbuf, vbar);
  ksum_kernel<<<dim3(HH, 40), 256, 0, stream>>>(phik, ksum);

  // 4) ktv (split-K) + transpose-convert to bf16
  ktv_kernel<<<dim3(DD / 16, HH, 4), 256, 0, stream>>>(phik, vbuf, ktvf);
  ktvT_kernel<<<(HH * DD * DD) / 256, 256, 0, stream>>>(ktvf, ktvT);

  // 5) final output
  out_kernel<<<NN / 16, 256, 0, stream>>>(phiq, vbar, ktvT, ksum, vmapbf, vmap_b, out);
}
